// MultiHeadAttention_49924699849253
// MI455X (gfx1250) — matrix operand order repaired and checked by reading, not compiled, not run on hardware
//
#include <hip/hip_runtime.h>
#include <stdint.h>

#define DEV __device__ __forceinline__

typedef __attribute__((ext_vector_type(16))) __bf16 v16bf;
typedef __attribute__((ext_vector_type(8)))  float  v8f;

constexpr int Bn  = 2;
constexpr int Sn  = 2048;
constexpr int Dm  = 1024;
constexpr int Hn  = 16;
constexpr int DKn = 64;
constexpr int Mn  = Bn * Sn;   // 4096

// ---------- bf16 helpers (storage type: uint16_t) ----------
DEV uint16_t f2bf(float f) {
  union { float f; uint32_t u; } a; a.f = f;
  uint32_t r = a.u + 0x7FFFu + ((a.u >> 16) & 1u);   // round-to-nearest-even
  return (uint16_t)(r >> 16);
}

union FragU { v16bf v; uint4 q[2]; };

// A fragment 16x32 bf16 (wave32): lane -> M=lane&15; 16 contiguous K at (lane>>4)*16
DEV v16bf load_a_frag(const uint16_t* base, int stride, int lane) {
  int m = lane & 15, bk = (lane >> 4) * 16;
  const uint16_t* r = base + m * stride + bk;
  FragU u;
  u.q[0] = *(const uint4*)(r);
  u.q[1] = *(const uint4*)(r + 8);
  return u.v;
}

// B fragment 32x16 bf16 (wave32): lane -> N=lane&15; 16 contiguous K at (lane>>4)*16
DEV v16bf load_b_frag(const uint16_t* base, int stride, int lane) {
  int n = lane & 15, kb = (lane >> 4) * 16;
  const uint16_t* r = base + n * stride + kb;
  FragU u;
  u.q[0] = ((const uint4*)r)[0];
  u.q[1] = ((const uint4*)r)[1];
  return u.v;
}

DEV v8f wmma_bf16(v16bf a, v16bf b, v8f c) {
  return __builtin_amdgcn_wmma_f32_16x16x32_bf16(false, a, false, b, (short)0, c,
                                                 false, false);
}

DEV v8f zero8() { v8f z = {0.f,0.f,0.f,0.f,0.f,0.f,0.f,0.f}; return z; }

// ---------- async global -> LDS copy (CDNA5, tracked by ASYNCcnt) ----------
DEV void async_copy16(uint32_t lds_addr, const uint16_t* g) {
  asm volatile("global_load_async_to_lds_b128 %0, %1, off"
               :: "v"(lds_addr), "v"((uint64_t)(uintptr_t)g)
               : "memory");
}
DEV void wait_async() {
  asm volatile("s_wait_asynccnt 0x0" ::: "memory");
}
// copy `n16` 16-byte chunks of one row
template <int N16>
DEV void stage_row(uint32_t lds_addr, const uint16_t* g) {
#pragma unroll
  for (int j = 0; j < N16; j++)
    async_copy16(lds_addr + 16u * j, g + 8 * j);
}

// ---------- elementwise fp32 -> bf16 ----------
__global__ void cvt_bf16(const float* __restrict__ src, uint16_t* __restrict__ dst,
                         int n) {
  int i = blockIdx.x * blockDim.x + threadIdx.x;
  if (i < n) dst[i] = f2bf(src[i]);
}

// ---------- weight transpose: w[K,N] fp32 -> wt[N,K] bf16 (K=N=Dm) ----------
__global__ __launch_bounds__(256)
void wtrans(const float* __restrict__ w, uint16_t* __restrict__ wt) {
  __shared__ float t[32][33];
  int n0 = blockIdx.x * 32, k0 = blockIdx.y * 32;
  int tx = threadIdx.x & 31, ty = threadIdx.x >> 5;   // 8 row-groups
  for (int i = ty; i < 32; i += 8)
    t[i][tx] = w[(size_t)(k0 + i) * Dm + n0 + tx];
  __syncthreads();
  for (int i = ty; i < 32; i += 8)
    wt[(size_t)(n0 + i) * Dm + k0 + tx] = f2bf(t[tx][i]);
}

// ---------- WMMA GEMM: A[M,K]bf16 @ Bt[N,K]bf16^T + bias ----------
// 128x128 tile, k-step 64, double-buffered LDS filled by async copies.
// MODE 0: dst bf16, head-split [b,h,s,dk]            (Q, K projections)
// MODE 1: dst bf16, head-split transposed [b,h,dk,s] (V projection)
// MODE 2: dst fp32 row-major [M,N]                   (output projection)
template <int MODE>
__global__ __launch_bounds__(256)
void gemm_bf16(const uint16_t* __restrict__ A, const uint16_t* __restrict__ Bt,
               const float* __restrict__ bias, void* __restrict__ dst,
               int M, int N, int K) {
  __shared__ uint16_t sA[2][128][64];   // 2 x 16 KB
  __shared__ uint16_t sB[2][128][64];   // 2 x 16 KB
  int tid = threadIdx.x, lane = tid & 31, wave = tid >> 5;
  int wr = wave >> 2, wc = wave & 3;                 // 2x4 wave grid
  int m0 = blockIdx.y * 128, n0 = blockIdx.x * 128;

  v8f acc[4][2];
  for (int i = 0; i < 4; i++)
    for (int j = 0; j < 2; j++) acc[i][j] = zero8();

  int lr = tid >> 1;                 // 0..127 (staging row)
  int lc = (tid & 1) * 32;           // 0 or 32 (elements, 64 B half-row)

  auto stage = [&](int buf, int k0) {
    stage_row<4>((uint32_t)(uintptr_t)&sA[buf][lr][lc],
                 A + (size_t)(m0 + lr) * K + k0 + lc);
    stage_row<4>((uint32_t)(uintptr_t)&sB[buf][lr][lc],
                 Bt + (size_t)(n0 + lr) * K + k0 + lc);
  };

  const int NK = K / 64;
  stage(0, 0);
  wait_async();
  __syncthreads();

  for (int it = 0; it < NK; it++) {
    int cur = it & 1;
    if (it + 1 < NK) stage(cur ^ 1, (it + 1) * 64);

#pragma unroll
    for (int kk = 0; kk < 2; kk++) {
      v16bf af[4], bfr[2];
#pragma unroll
      for (int mt = 0; mt < 4; mt++)
        af[mt] = load_a_frag(&sA[cur][wr * 64 + mt * 16][kk * 32], 64, lane);
#pragma unroll
      for (int nt = 0; nt < 2; nt++)
        bfr[nt] = load_b_frag(&sB[cur][wc * 32 + nt * 16][kk * 32], 64, lane);
#pragma unroll
      for (int mt = 0; mt < 4; mt++)
#pragma unroll
        for (int nt = 0; nt < 2; nt++)
          acc[mt][nt] = wmma_bf16(af[mt], bfr[nt], acc[mt][nt]);
    }

    wait_async();
    __syncthreads();
  }

  int rbase = (lane >> 4) * 8;       // C/D layout: M = r + 8*(lane>=16)
#pragma unroll
  for (int mt = 0; mt < 4; mt++)
#pragma unroll
    for (int nt = 0; nt < 2; nt++) {
      int gn = n0 + wc * 32 + nt * 16 + (lane & 15);
      float bv = bias[gn];
#pragma unroll
      for (int r = 0; r < 8; r++) {
        int gm = m0 + wr * 64 + mt * 16 + rbase + r;
        float v = acc[mt][nt][r] + bv;
        if (MODE == 2) {
          ((float*)dst)[(size_t)gm * N + gn] = v;
        } else {
          int bb = gm / Sn, ss = gm % Sn, hh = gn / DKn, dk = gn % DKn;
          size_t off = (MODE == 0)
            ? ((size_t)(bb * Hn + hh) * Sn + ss) * DKn + dk
            : ((size_t)(bb * Hn + hh) * DKn + dk) * Sn + ss;
          ((uint16_t*)dst)[off] = f2bf(v);
        }
      }
    }
}

// ---------- flash attention: 1 WG = 64 query rows of one (b,h); 4 waves ----------
__global__ __launch_bounds__(128)
void attention_kernel(const uint16_t* __restrict__ Q,    // [B,H,S,DK]
                      const uint16_t* __restrict__ Km,   // [B,H,S,DK]
                      const uint16_t* __restrict__ Vt,   // [B,H,DK,S]
                      const int* __restrict__ mask,      // [B,1,S,S]
                      uint16_t* __restrict__ O) {        // [B,S,D] bf16
  __shared__ uint16_t sQ[64][DKn];          // 8 KB
  __shared__ uint16_t sK[2][64][DKn];       // 16 KB  rows = key
  __shared__ uint16_t sV[2][DKn][64];       // 16 KB  rows = d, cols = key
  __shared__ uint16_t sP[4][16][64];        // 8 KB   per-wave P staging

  int tid = threadIdx.x, lane = tid & 31, wave = tid >> 5;
  int q0 = blockIdx.x * 64;
  int bh = blockIdx.y, b = bh / Hn, h = bh % Hn;
  int rbase = (lane >> 4) * 8;

  // stage K/V block kb into buffer buf (no intra-wave divergence: 64 = 2 waves)
  auto stageKV = [&](int buf, int kb) {
    int row = tid & 63;
    if (tid < 64) {
      stage_row<8>((uint32_t)(uintptr_t)&sK[buf][row][0],
                   Km + (((size_t)bh) * Sn + kb * 64 + row) * DKn);
    } else {
      stage_row<8>((uint32_t)(uintptr_t)&sV[buf][row][0],
                   Vt + (((size_t)bh) * DKn + row) * Sn + kb * 64);
    }
  };

  // load Q tile (64 x 64 bf16) + first K/V block
  {
    int row = tid >> 1, ho = (tid & 1) * 32;     // 32 elems = 64 B
    stage_row<4>((uint32_t)(uintptr_t)&sQ[row][ho],
                 Q + (((size_t)bh) * Sn + q0 + row) * DKn + ho);
  }
  stageKV(0, 0);
  wait_async();
  __syncthreads();

  v16bf aq0 = load_a_frag(&sQ[wave * 16][0],  DKn, lane);
  v16bf aq1 = load_a_frag(&sQ[wave * 16][32], DKn, lane);

  v8f o[4];
  for (int i = 0; i < 4; i++) o[i] = zero8();
  float mrow[8], lrow[8];
#pragma unroll
  for (int r = 0; r < 8; r++) { mrow[r] = -3.0e38f; lrow[r] = 0.0f; }

  const int* mb = mask + (size_t)b * Sn * Sn;
  const int NB = Sn / 64;

  for (int kb = 0; kb < NB; kb++) {
    int cur = kb & 1;
    if (kb + 1 < NB) stageKV(cur ^ 1, kb + 1);

    // scores S = Q @ K^T  (per wave: 16 x 64)
    v8f c[4];
#pragma unroll
    for (int nt = 0; nt < 4; nt++) {
      v16bf b0 = load_b_frag(&sK[cur][nt * 16][0],  DKn, lane);
      v16bf b1 = load_b_frag(&sK[cur][nt * 16][32], DKn, lane);
      v8f cc = zero8();
      cc = wmma_bf16(aq0, b0, cc);
      cc = wmma_bf16(aq1, b1, cc);
      c[nt] = cc;
    }

    // scale + mask (masked positions become 1e-9, still inside softmax)
    float sc[4][8];
#pragma unroll
    for (int nt = 0; nt < 4; nt++) {
      int kcol = kb * 64 + nt * 16 + (lane & 15);
#pragma unroll
      for (int r = 0; r < 8; r++) {
        int qrow = q0 + wave * 16 + rbase + r;
        int mv = mb[(size_t)qrow * Sn + kcol];
        float s = c[nt][r] * 0.125f;
        sc[nt][r] = (mv == 0) ? 1e-9f : s;
      }
    }

    // online softmax update (row reductions stay inside 16-lane groups)
#pragma unroll
    for (int r = 0; r < 8; r++) {
      float bm = fmaxf(fmaxf(sc[0][r], sc[1][r]), fmaxf(sc[2][r], sc[3][r]));
      bm = fmaxf(bm, __shfl_xor(bm, 1));
      bm = fmaxf(bm, __shfl_xor(bm, 2));
      bm = fmaxf(bm, __shfl_xor(bm, 4));
      bm = fmaxf(bm, __shfl_xor(bm, 8));
      float mnew = fmaxf(mrow[r], bm);
      float corr = __expf(mrow[r] - mnew);
      mrow[r] = mnew;
      lrow[r] *= corr;
#pragma unroll
      for (int nt = 0; nt < 4; nt++) o[nt][r] *= corr;
      float ps = 0.0f;
#pragma unroll
      for (int nt = 0; nt < 4; nt++) {
        float p = __expf(sc[nt][r] - mnew);
        ps += p;
        sP[wave][rbase + r][nt * 16 + (lane & 15)] = f2bf(p);
      }
      ps += __shfl_xor(ps, 1);
      ps += __shfl_xor(ps, 2);
      ps += __shfl_xor(ps, 4);
      ps += __shfl_xor(ps, 8);
      lrow[r] += ps;
    }

    // O += P @ V  (P re-read from LDS in A-fragment layout)
    v16bf ap0 = load_a_frag(&sP[wave][0][0],  64, lane);
    v16bf ap1 = load_a_frag(&sP[wave][0][32], 64, lane);
#pragma unroll
    for (int nt = 0; nt < 4; nt++) {
      v16bf b0 = load_b_frag(&sV[cur][nt * 16][0],  64, lane);
      v16bf b1 = load_b_frag(&sV[cur][nt * 16][32], 64, lane);
      o[nt] = wmma_bf16(ap0, b0, o[nt]);
      o[nt] = wmma_bf16(ap1, b1, o[nt]);
    }

    wait_async();
    __syncthreads();
  }

  // epilogue: normalize and store bf16 into [B,S,D] (merged heads)
#pragma unroll
  for (int nt = 0; nt < 4; nt++) {
    int dcol = h * DKn + nt * 16 + (lane & 15);
#pragma unroll
    for (int r = 0; r < 8; r++) {
      int gm = q0 + wave * 16 + rbase + r;
      float v = o[nt][r] / lrow[r];
      O[((size_t)b * Sn + gm) * Dm + dcol] = f2bf(v);
    }
  }
}

extern "C" void kernel_launch(void* const* d_in, const int* in_sizes, int n_in,
                              void* d_out, int out_size, void* d_ws, size_t ws_size,
                              hipStream_t stream) {
  (void)in_sizes; (void)n_in; (void)out_size; (void)ws_size;
  const float* q    = (const float*)d_in[0];
  const float* k    = (const float*)d_in[1];
  const float* v    = (const float*)d_in[2];
  const int*   mask = (const int*)d_in[3];
  const float* wq   = (const float*)d_in[4];
  const float* bq   = (const float*)d_in[5];
  const float* wk   = (const float*)d_in[6];
  const float* bk   = (const float*)d_in[7];
  const float* wv   = (const float*)d_in[8];
  const float* bv   = (const float*)d_in[9];
  const float* wo   = (const float*)d_in[10];
  const float* bo   = (const float*)d_in[11];

  char* ws = (char*)d_ws;
  const size_t MB = 1024u * 1024u;
  uint16_t* qb  = (uint16_t*)(ws + 0 * MB);    // [M,D] bf16
  uint16_t* kb_ = (uint16_t*)(ws + 8 * MB);
  uint16_t* vb  = (uint16_t*)(ws + 16 * MB);
  uint16_t* wqT = (uint16_t*)(ws + 24 * MB);   // [N,K] bf16
  uint16_t* wkT = (uint16_t*)(ws + 26 * MB);
  uint16_t* wvT = (uint16_t*)(ws + 28 * MB);
  uint16_t* woT = (uint16_t*)(ws + 30 * MB);
  uint16_t* Qh  = (uint16_t*)(ws + 32 * MB);   // [B,H,S,DK]
  uint16_t* Kh  = (uint16_t*)(ws + 40 * MB);   // [B,H,S,DK]
  uint16_t* Vth = (uint16_t*)(ws + 48 * MB);   // [B,H,DK,S]
  uint16_t* Ab  = (uint16_t*)(ws + 56 * MB);   // [B,S,D] bf16 attn output

  int n = Mn * Dm;   // 4,194,304
  cvt_bf16<<<n / 256, 256, 0, stream>>>(q, qb, n);
  cvt_bf16<<<n / 256, 256, 0, stream>>>(k, kb_, n);
  cvt_bf16<<<n / 256, 256, 0, stream>>>(v, vb, n);

  dim3 tg(Dm / 32, Dm / 32);
  wtrans<<<tg, 256, 0, stream>>>(wq, wqT);
  wtrans<<<tg, 256, 0, stream>>>(wk, wkT);
  wtrans<<<tg, 256, 0, stream>>>(wv, wvT);
  wtrans<<<tg, 256, 0, stream>>>(wo, woT);

  dim3 gg(Dm / 128, Mn / 128);   // (8, 32)
  gemm_bf16<0><<<gg, 256, 0, stream>>>(qb,  wqT, bq, Qh,  Mn, Dm, Dm);
  gemm_bf16<0><<<gg, 256, 0, stream>>>(kb_, wkT, bk, Kh,  Mn, Dm, Dm);
  gemm_bf16<1><<<gg, 256, 0, stream>>>(vb,  wvT, bv, Vth, Mn, Dm, Dm);

  dim3 ga(Sn / 64, Bn * Hn);     // (32, 32)
  attention_kernel<<<ga, 128, 0, stream>>>(Qh, Kh, Vth, mask, Ab);

  gemm_bf16<2><<<gg, 256, 0, stream>>>(Ab, woT, bo, d_out, Mn, Dm, Dm);
}
